// Attention_2130303779056
// MI455X (gfx1250) — matrix operand order repaired and checked by reading, not compiled, not run on hardware
//
#include <hip/hip_runtime.h>

// ---- problem constants (match reference) ----
#define ATT_B  2
#define ATT_H  16
#define ATT_S  2048
#define ATT_D  64

#define BM     128           // rows of Q per workgroup (8 waves x 16)
#define BN     64            // KV columns per iteration
#define NWAVES 8
#define NTHR   (NWAVES * 32)
#define KS     72            // padded LDS row stride (bf16 elems) for K tile [t][d]
#define VS     72            // padded LDS row stride for V tile [d][t]
#define PS     72            // padded LDS row stride for P staging [m][t]

typedef __attribute__((ext_vector_type(16))) __bf16 v16bf;
typedef __attribute__((ext_vector_type(8)))  float  v8f;

union FragBF {
    unsigned short u[16];
    unsigned int   d[8];
    v16bf          v;
};

// round two f32 to bf16 (RNE) and pack into one dword: lo -> [15:0], hi -> [31:16]
__device__ __forceinline__ unsigned int pack_bf16x2(float lo, float hi) {
    unsigned int a = __float_as_uint(lo);
    unsigned int b = __float_as_uint(hi);
    a += 0x7fffu + ((a >> 16) & 1u);
    b += 0x7fffu + ((b >> 16) & 1u);
    return __builtin_amdgcn_perm(b, a, 0x07060302u);   // {b[3],b[2],a[3],a[2]}
}

__device__ __forceinline__ unsigned short f2bf(float f) {
    unsigned int u = __float_as_uint(f);
    u += 0x7fffu + ((u >> 16) & 1u);
    return (unsigned short)(u >> 16);
}

// xor-butterfly step within 16-lane groups (VALU path via v_permlane16_b32)
__device__ __forceinline__ float xor16(float x, int sel0, int sel1) {
#if __has_builtin(__builtin_amdgcn_permlane16)
    int i = __float_as_int(x);
    i = __builtin_amdgcn_permlane16(i, i, sel0, sel1, false, false);
    return __int_as_float(i);
#else
    (void)sel0; (void)sel1;
    return x;
#endif
}

__device__ __forceinline__ float rowmax16(float t) {
#if __has_builtin(__builtin_amdgcn_permlane16)
    t = fmaxf(t, xor16(t, 0x67452301, (int)0xEFCDAB89));  // xor 1
    t = fmaxf(t, xor16(t, 0x54761032, (int)0xDCFE98BA));  // xor 2
    t = fmaxf(t, xor16(t, 0x32107654, (int)0xBA98FEDC));  // xor 4
    t = fmaxf(t, xor16(t, (int)0xFEDCBA98, 0x76543210));  // xor 8
#else
#pragma unroll
    for (int off = 8; off >= 1; off >>= 1)
        t = fmaxf(t, __shfl_xor(t, off, 16));
#endif
    return t;
}

// A-matrix (16x32 bf16) K-pair base for VGPR i, given lane half
__device__ __forceinline__ int kbA(int i, int half) {
    return ((i < 4) ? (2 * i) : (16 + 2 * (i - 4))) + half * 8;
}
// B-matrix (32x16 bf16) K-pair base for VGPR i, given lane half
__device__ __forceinline__ int kbB(int i, int half) {
    return ((i < 4) ? (2 * i) : (16 + 2 * (i - 4))) + half * 8;
}

__global__ __launch_bounds__(NTHR, 2)
void fattn_kernel(const float* __restrict__ q,
                  const float* __restrict__ k,
                  const float* __restrict__ v,
                  float* __restrict__ out)
{
    const int tid  = threadIdx.x;
    const int wave = tid >> 5;
    const int lane = tid & 31;
    const int half = lane >> 4;     // 0: lanes 0-15, 1: lanes 16-31
    const int ln   = lane & 15;

    const int wg   = blockIdx.x;
    const int nmb  = ATT_S / BM;    // 16 row blocks
    const int bh   = wg / nmb;      // B*H = 32 heads
    const int mblk = wg % nmb;
    const int m0   = mblk * BM;

    const size_t base = (size_t)bh * ATT_S * ATT_D;   // same flat base for q, k, v, out

    __shared__ unsigned short sK[BN * KS];            // K tile transposed: [t][d]
    __shared__ unsigned short sV[ATT_D * VS];         // V tile transposed: [d][t]
    __shared__ unsigned short sP[NWAVES][16 * PS];    // per-wave P staging: [m][t]

    // ---- all-ones B fragment (bf16 1.0) for row-sum WMMA ----
    FragBF onesB;
#pragma unroll
    for (int i = 0; i < 8; ++i) onesB.d[i] = 0x3F803F80u;

    // ---- load Q fragments (A layout, bf16), held for whole sweep ----
    FragBF aQ[2];
    {
        const float* qrow = q + base + (size_t)(m0 + wave * 16 + ln) * ATT_D;
#pragma unroll
        for (int h = 0; h < 2; ++h) {
#pragma unroll
            for (int i = 0; i < 8; ++i) {
                int kb = kbA(i, half) + h * 32;
                aQ[h].u[2 * i]     = f2bf(qrow[kb]);
                aQ[h].u[2 * i + 1] = f2bf(qrow[kb + 1]);
            }
        }
    }

    // ---- running softmax state (lanes 0-15: rows r, lanes 16-31: rows r+8) ----
    float mrun[8], lrun[8];
#pragma unroll
    for (int r = 0; r < 8; ++r) { mrun[r] = -3.0e38f; lrun[r] = 0.0f; }
    v8f accO[4] = {};   // O accumulators, 4 x (16 cols of D)

    for (int t0 = 0; t0 < ATT_S; t0 += BN) {
        __syncthreads();
        // ---- stage K tile: global [d][t] -> LDS [t][d], one bf16 pair per store ----
        // pair p: t = p & 63 (coalesced across lanes), d-pair = p >> 6
#pragma unroll
        for (int r = 0; r < (BN * ATT_D) / 2 / NTHR; ++r) {
            int p  = tid + r * NTHR;
            int t  = p & 63, d2 = p >> 6;
            const float* kp = &k[base + (size_t)(2 * d2) * ATT_S + t0 + t];
            *(unsigned int*)&sK[t * KS + 2 * d2] = pack_bf16x2(kp[0], kp[ATT_S]);
        }
        // ---- stage V tile: global [t][d] -> LDS [d][t], one bf16 pair per store ----
#pragma unroll
        for (int r = 0; r < (BN * ATT_D) / 2 / NTHR; ++r) {
            int p  = tid + r * NTHR;
            int d  = p & 63, t2 = p >> 6;
            const float* vp = &v[base + (size_t)(t0 + 2 * t2) * ATT_D + d];
            *(unsigned int*)&sV[d * VS + 2 * t2] = pack_bf16x2(vp[0], vp[ATT_D]);
        }
        if (t0 + BN < ATT_S) {    // prefetch next tiles -> global_prefetch_b8
            __builtin_prefetch(&k[base + (size_t)(tid >> 6) * ATT_S + t0 + BN + (tid & 63)], 0, 0);
            __builtin_prefetch(&v[base + (size_t)(t0 + BN + (tid >> 6)) * ATT_D + (tid & 63)], 0, 0);
        }
        __syncthreads();

        // ---- S = Q @ K^T : 4 N-subtiles x 2 K-halves of WMMA ----
        v8f sc[4];
#pragma unroll
        for (int j = 0; j < 4; ++j) {
            v8f c = {};
#pragma unroll
            for (int h = 0; h < 2; ++h) {
                FragBF bK;
                int col = j * 16 + ln;
#pragma unroll
                for (int i = 0; i < 8; ++i) {
                    int kb = kbB(i, half) + h * 32;            // even -> 4B aligned pair
                    bK.d[i] = *(const unsigned int*)&sK[col * KS + kb];
                }
                c = __builtin_amdgcn_wmma_f32_16x16x32_bf16(
                        false, aQ[h].v, false, bK.v, (short)0, c, false, false);
            }
            sc[j] = c;
        }

        // ---- row max (permlane16 butterfly, VALU pipe) ----
        float mnew[8], scl[8];
#pragma unroll
        for (int r = 0; r < 8; ++r) {
            float t = fmaxf(fmaxf(sc[0][r], sc[1][r]), fmaxf(sc[2][r], sc[3][r]));
            t = rowmax16(t);
            mnew[r] = fmaxf(mrun[r], t);
            scl[r]  = __expf(mrun[r] - mnew[r]);
            mrun[r] = mnew[r];
        }

        // ---- P = exp(S - mnew); stage to LDS for layout transpose ----
#pragma unroll
        for (int j = 0; j < 4; ++j)
#pragma unroll
            for (int r = 0; r < 8; ++r) {
                float p = __expf(sc[j][r] - mnew[r]);
                sP[wave][(r + half * 8) * PS + j * 16 + ln] = f2bf(p);
            }
        asm volatile("s_wait_dscnt 0x0" ::: "memory");   // wave-local LDS RAW fence

        FragBF aP[2];
#pragma unroll
        for (int h = 0; h < 2; ++h)
#pragma unroll
            for (int i = 0; i < 8; ++i) {
                int kb = kbA(i, half) + h * 32;
                aP[h].d[i] = *(const unsigned int*)&sP[wave][ln * PS + kb];
            }

        // ---- row sums via WMMA: rs = P @ ones(64x16); C-layout row stats ----
        {
            v8f rs = {};
            rs = __builtin_amdgcn_wmma_f32_16x16x32_bf16(
                     false, aP[0].v, false, onesB.v, (short)0, rs, false, false);
            rs = __builtin_amdgcn_wmma_f32_16x16x32_bf16(
                     false, aP[1].v, false, onesB.v, (short)0, rs, false, false);
#pragma unroll
            for (int r = 0; r < 8; ++r)
                lrun[r] = lrun[r] * scl[r] + rs[r];
        }

        // ---- rescale O accumulators ----
#pragma unroll
        for (int j = 0; j < 4; ++j)
#pragma unroll
            for (int r = 0; r < 8; ++r)
                accO[j][r] *= scl[r];

        // ---- O += P @ V ----
#pragma unroll
        for (int j = 0; j < 4; ++j) {
#pragma unroll
            for (int h = 0; h < 2; ++h) {
                FragBF bV;
                int col = j * 16 + ln;
#pragma unroll
                for (int i = 0; i < 8; ++i) {
                    int kb = kbB(i, half) + h * 32;
                    bV.d[i] = *(const unsigned int*)&sV[col * VS + kb];
                }
                accO[j] = __builtin_amdgcn_wmma_f32_16x16x32_bf16(
                              false, aP[h].v, false, bV.v, (short)0, accO[j], false, false);
            }
        }
    }

    // ---- epilogue: normalize and store ----
    float inv[8];
#pragma unroll
    for (int r = 0; r < 8; ++r) inv[r] = 1.0f / lrun[r];
#pragma unroll
    for (int j = 0; j < 4; ++j) {
#pragma unroll
        for (int r = 0; r < 8; ++r) {
            int row = m0 + wave * 16 + r + half * 8;
            out[base + (size_t)row * ATT_D + j * 16 + ln] = accO[j][r] * inv[r];
        }
    }
}

extern "C" void kernel_launch(void* const* d_in, const int* in_sizes, int n_in,
                              void* d_out, int out_size, void* d_ws, size_t ws_size,
                              hipStream_t stream) {
    (void)in_sizes; (void)n_in; (void)d_ws; (void)ws_size; (void)out_size;
    const float* q = (const float*)d_in[0];
    const float* k = (const float*)d_in[1];
    const float* v = (const float*)d_in[2];
    float* out = (float*)d_out;

    dim3 grid(ATT_B * ATT_H * (ATT_S / BM));   // 512 workgroups
    dim3 block(NTHR);                          // 8 waves (wave32)
    hipLaunchKernelGGL(fattn_kernel, grid, block, 0, stream, q, k, v, out);
}
